// VectorQuantizer_76416058131071
// MI455X (gfx1250) — matrix operand order repaired and checked by reading, not compiled, not run on hardware
//
#include <hip/hip_runtime.h>
#include <hip/hip_bf16.h>

typedef __attribute__((ext_vector_type(16))) __bf16 v16bf;
typedef __attribute__((ext_vector_type(8)))  float  v8f;

#define N_TOK   16384   // B*T
#define D_IN    768
#define D_MOD   512
#define K_CB    8192

// Native bf16 convert (gfx1250 has hardware bf16); backend selects v_cvt.
__device__ __forceinline__ __bf16 f2bf(float f) { return (__bf16)f; }

// ---------------- prep: zero accumulators ----------------
__global__ void vq_zero(int* __restrict__ counts, float* __restrict__ sse) {
    int t = blockIdx.x * blockDim.x + threadIdx.x;
    if (t < K_CB) counts[t] = 0;
    if (t == 0)  *sse = 0.f;
}

// ---------------- prep: W_proj -> bf16 B-layout ----------------
// B-layout (16-bit, 32x16): lane<16: col=lane,     h<8 -> K=h,    h>=8 -> K=16+(h-8)
//                           lane>=16: col=lane-16, h<8 -> K=8+h,  h>=8 -> K=24+(h-8)
__global__ void vq_convW(const float* __restrict__ W, __bf16* __restrict__ Wsw) {
    int t = blockIdx.x * blockDim.x + threadIdx.x;          // (DIN/32)*(D/16)*32 = 24576
    if (t >= 24 * 32 * 32) return;
    int lane = t & 31, tile = t >> 5;
    int dc = tile >> 5, ct = tile & 31;
    int col = lane & 15;
    int koff = (lane < 16) ? 0 : 8;
    v16bf v;
#pragma unroll
    for (int h = 0; h < 16; ++h)
        v[h] = f2bf(W[(size_t)(dc * 32 + koff + h + (h & 8)) * D_MOD + ct * 16 + col]);
    *(v16bf*)(Wsw + (size_t)t * 16) = v;
}

// ---------------- prep: codebook^T -> bf16 B-layout ----------------
__global__ void vq_convCB(const float* __restrict__ cb, __bf16* __restrict__ cbsw) {
    int t = blockIdx.x * blockDim.x + threadIdx.x;          // (K/16)*(D/32)*32 = 262144
    int lane = t & 31, tile = t >> 5;                        // tile = kt*16 + dc
    int kt = tile >> 4, dc = tile & 15;
    int cw = kt * 16 + (lane & 15);
    int koff = (lane < 16) ? 0 : 8;
    const float* p = cb + (size_t)cw * D_MOD + dc * 32 + koff;
    v16bf v;
#pragma unroll
    for (int h = 0; h < 16; ++h) v[h] = f2bf(p[h + (h & 8)]);
    *(v16bf*)(cbsw + (size_t)t * 16) = v;
}

// ---------------- prep: ||c_k||^2 (f32, exact ranking bias) ----------------
__global__ void vq_cnorm(const float* __restrict__ cb, float* __restrict__ cnorm) {
    __shared__ float red[128];
    int k = blockIdx.x, t = threadIdx.x;
    float4 v = ((const float4*)cb)[(size_t)k * 128 + t];
    red[t] = v.x * v.x + v.y * v.y + v.z * v.z + v.w * v.w;
    __syncthreads();
    for (int o = 64; o; o >>= 1) { if (t < o) red[t] += red[t + o]; __syncthreads(); }
    if (!t) cnorm[k] = red[0];
}

// ---------------- projection GEMM: inputs = x @ W + b (bf16 WMMA, f32 accum) ----------------
// A-layout (16-bit, 16x32): lane<16: row=lane,     h<8 -> K=h,    h>=8 -> K=16+(h-8)
//                           lane>=16: row=lane-16, h<8 -> K=8+h,  h>=8 -> K=24+(h-8)
__global__ void vq_proj(const float* __restrict__ x, const __bf16* __restrict__ Wsw,
                        const float* __restrict__ bproj, float* __restrict__ inputs) {
    const int nt   = blockIdx.x;                 // token tile 0..1023
    const int w    = threadIdx.x >> 5;           // wave 0..7
    const int lane = threadIdx.x & 31;
    const int ct   = blockIdx.y * 8 + w;         // col tile 0..31
    const int row  = nt * 16 + (lane & 15);
    const int koff = (lane < 16) ? 0 : 8;
    v8f acc = {};
    for (int dc = 0; dc < 24; ++dc) {            // D_IN / 32
        const float* ap = x + (size_t)row * D_IN + dc * 32 + koff;
        v16bf a;
#pragma unroll
        for (int h = 0; h < 8; ++h) a[h] = f2bf(ap[h]);
#pragma unroll
        for (int h = 0; h < 8; ++h) a[8 + h] = f2bf(ap[16 + h]);
        v16bf b = *(const v16bf*)(Wsw + ((size_t)(dc * 32 + ct) * 32 + lane) * 16);
        acc = __builtin_amdgcn_wmma_f32_16x16x32_bf16(false, a, false, b,
                                                      (short)0, acc, false, false);
    }
    const int col = lane & 15;
    const float bb = bproj[ct * 16 + col];
    const int rbase = (lane < 16) ? 0 : 8;       // C layout: VGPR r -> M = r (+8 hi lanes)
#pragma unroll
    for (int r = 0; r < 8; ++r)
        inputs[(size_t)(nt * 16 + rbase + r) * D_MOD + ct * 16 + col] = acc[r] + bb;
}

// ---------------- re-swizzle inputs -> bf16 A-layout for the distance GEMM ----------------
__global__ void vq_swzIn(const float* __restrict__ inputs, __bf16* __restrict__ insw) {
    int t = blockIdx.x * blockDim.x + threadIdx.x;          // (N/16)*(D/32)*32 = 524288
    int lane = t & 31, tile = t >> 5;                        // tile = nt*16 + dc
    int nt = tile >> 4, dc = tile & 15;
    int row = nt * 16 + (lane & 15);
    int koff = (lane < 16) ? 0 : 8;
    const float* p = inputs + (size_t)row * D_MOD + dc * 32 + koff;
    v16bf v;
#pragma unroll
    for (int h = 0; h < 8; ++h) v[h] = f2bf(p[h]);
#pragma unroll
    for (int h = 0; h < 8; ++h) v[8 + h] = f2bf(p[16 + h]);
    *(v16bf*)(insw + (size_t)t * 16) = v;
}

// ---------------- fused distance GEMM + argmin ----------------
// Block = 128 tokens (8 waves x 16-token register A tiles); all waves sweep the
// SAME codeword tile, staged in LDS double buffers -> 8x less codebook L2 traffic.
// score = ||c||^2 - 2*f.c  (||f||^2 constant per row -> dropped for argmin)
__global__ void __launch_bounds__(256)
vq_dist(const __bf16* __restrict__ insw, const __bf16* __restrict__ cbsw,
        const float* __restrict__ cnorm, int* __restrict__ tok_i,
        float* __restrict__ out_tok, int* __restrict__ counts) {
    const int w    = threadIdx.x >> 5;
    const int lane = threadIdx.x & 31;
    const int col  = lane & 15;
    const int nt   = blockIdx.x * 8 + w;                      // this wave's token tile

    // A tile (16 tokens x 512 dims) in registers: 512x reuse
    v16bf A[16];
    const __bf16* ab = insw + (size_t)nt * 16 * 32 * 16;
#pragma unroll
    for (int dc = 0; dc < 16; ++dc)
        A[dc] = *(const v16bf*)(ab + ((size_t)dc * 32 + lane) * 16);

    float bs[8]; int bi[8];
#pragma unroll
    for (int r = 0; r < 8; ++r) { bs[r] = 3.4e38f; bi[r] = 0; }

    // 2 x 16KB LDS double buffer for the shared B tile (16 dc x 32 lanes x 16 hw)
    __shared__ __align__(16) __bf16 Bsh[2][16 * 32 * 16];
    uint4 cr0, cr1, cr2, cr3;                                 // 64B staging regs/thread

    { // prime buffer 0
        const uint4* s = (const uint4*)(cbsw) + (size_t)threadIdx.x * 4;
        cr0 = s[0]; cr1 = s[1]; cr2 = s[2]; cr3 = s[3];
        uint4* d = (uint4*)(&Bsh[0][0]) + (size_t)threadIdx.x * 4;
        d[0] = cr0; d[1] = cr1; d[2] = cr2; d[3] = cr3;
    }

    for (int kt = 0; kt < K_CB / 16; ++kt) {
        const int cur = kt & 1;
        __syncthreads();                                      // buf[cur] ready / safe
        if (kt + 1 < K_CB / 16) {                             // global loads overlap WMMA
            const uint4* s = (const uint4*)(cbsw + (size_t)(kt + 1) * 16 * 32 * 16)
                             + (size_t)threadIdx.x * 4;
            cr0 = s[0]; cr1 = s[1]; cr2 = s[2]; cr3 = s[3];
        }
        v8f acc = {};
        const __bf16* bb = &Bsh[cur][0];
#pragma unroll
        for (int dc = 0; dc < 16; ++dc) {
            v16bf b = *(const v16bf*)(bb + (dc * 32 + lane) * 16);
            acc = __builtin_amdgcn_wmma_f32_16x16x32_bf16(false, A[dc], false, b,
                                                          (short)0, acc, false, false);
        }
        if (kt + 1 < K_CB / 16) {
            uint4* d = (uint4*)(&Bsh[cur ^ 1][0]) + (size_t)threadIdx.x * 4;
            d[0] = cr0; d[1] = cr1; d[2] = cr2; d[3] = cr3;
        }
        const float cn = cnorm[kt * 16 + col];
        const int idx  = kt * 16 + col;
#pragma unroll
        for (int r = 0; r < 8; ++r) {
            float s = fmaf(-2.f, acc[r], cn);
            if (s < bs[r] || (s == bs[r] && idx < bi[r])) { bs[r] = s; bi[r] = idx; }
        }
    }

    // butterfly min across the 16 lanes of each half-wave (wave32)
#pragma unroll
    for (int r = 0; r < 8; ++r) {
        float s = bs[r]; int i = bi[r];
        for (int off = 8; off; off >>= 1) {
            float os = __shfl_xor(s, off, 32);
            int   oi = __shfl_xor(i, off, 32);
            if (os < s || (os == s && oi < i)) { s = os; i = oi; }
        }
        bs[r] = s; bi[r] = i;
    }
    if (col == 0) {     // lane 0 -> rows 0..7, lane 16 -> rows 8..15
        int rbase = (lane < 16) ? 0 : 8;
#pragma unroll
        for (int r = 0; r < 8; ++r) {
            int n = nt * 16 + rbase + r;
            tok_i[n]   = bi[r];
            out_tok[n] = (float)bi[r];
            atomicAdd(&counts[bi[r]], 1);
        }
    }
}

// ---------------- gather quantized (== quantized_st) + SSE for losses ----------------
__global__ void vq_gather(const float* __restrict__ cb, const float* __restrict__ inputs,
                          const int* __restrict__ tok_i, float* __restrict__ outq,
                          float* __restrict__ sse) {
    __shared__ float red[128];
    int n = blockIdx.x, t = threadIdx.x;
    int idx = tok_i[n];
    float4 q  = ((const float4*)cb)[(size_t)idx * 128 + t];
    float4 in = ((const float4*)inputs)[(size_t)n * 128 + t];
    ((float4*)outq)[(size_t)n * 128 + t] = q;
    float dx = q.x - in.x, dy = q.y - in.y, dz = q.z - in.z, dw = q.w - in.w;
    red[t] = dx * dx + dy * dy + dz * dz + dw * dw;
    __syncthreads();
    for (int o = 64; o; o >>= 1) { if (t < o) red[t] += red[t + o]; __syncthreads(); }
    if (!t) atomicAdd(sse, red[0]);
}

// ---------------- vq_loss + perplexity ----------------
__global__ void vq_final(const int* __restrict__ counts, const float* __restrict__ sse,
                         float* __restrict__ out2) {
    __shared__ float red[1024];
    int t = threadIdx.x;
    float h = 0.f;
    for (int k = t; k < K_CB; k += 1024) {
        float p = (float)counts[k] * (1.f / (float)N_TOK);
        h -= p * logf(p + 1e-10f);
    }
    red[t] = h; __syncthreads();
    for (int o = 512; o; o >>= 1) { if (t < o) red[t] += red[t + o]; __syncthreads(); }
    if (!t) {
        out2[0] = 1.25f * (*sse) * (1.f / ((float)N_TOK * (float)D_MOD)); // (1+0.25)*MSE
        out2[1] = expf(red[0]);
    }
}

extern "C" void kernel_launch(void* const* d_in, const int* in_sizes, int n_in,
                              void* d_out, int out_size, void* d_ws, size_t ws_size,
                              hipStream_t stream) {
    const float* x     = (const float*)d_in[0];
    const float* W     = (const float*)d_in[1];
    const float* bproj = (const float*)d_in[2];
    const float* cb    = (const float*)d_in[3];

    float* out_q   = (float*)d_out;                        // N*D
    float* out_tok = out_q + (size_t)N_TOK * D_MOD;        // N
    float* out2    = out_tok + N_TOK;                      // vq_loss, perplexity

    char* p = (char*)d_ws;
    float*  inputs = (float*)p;  p += (size_t)N_TOK * D_MOD * 4;   // 33.5 MB f32 projections
    float*  cnorm  = (float*)p;  p += (size_t)K_CB * 4;
    int*    counts = (int*)p;    p += (size_t)K_CB * 4;
    float*  sse    = (float*)p;  p += 256;
    int*    tok_i  = (int*)p;    p += (size_t)N_TOK * 4;
    __bf16* Wsw    = (__bf16*)p; p += (size_t)D_IN * D_MOD * 2;    // 0.8 MB
    __bf16* cbsw   = (__bf16*)p; p += (size_t)K_CB * D_MOD * 2;    // 8.4 MB (L2-resident)
    __bf16* insw   = (__bf16*)p;                                   // 16.8 MB

    vq_zero  <<<32, 256, 0, stream>>>(counts, sse);
    vq_convW <<<96, 256, 0, stream>>>(W, Wsw);
    vq_convCB<<<1024, 256, 0, stream>>>(cb, cbsw);
    vq_cnorm <<<K_CB, 128, 0, stream>>>(cb, cnorm);
    vq_proj  <<<dim3(N_TOK / 16, 4), 256, 0, stream>>>(x, Wsw, bproj, inputs);
    vq_swzIn <<<2048, 256, 0, stream>>>(inputs, insw);
    vq_dist  <<<N_TOK / 128, 256, 0, stream>>>(insw, cbsw, cnorm, tok_i, out_tok, counts);
    vq_gather<<<N_TOK, 128, 0, stream>>>(cb, inputs, tok_i, out_q, sse);
    vq_final <<<1, 1024, 0, stream>>>(counts, sse, out2);
}
